// GPT_48859547959310
// MI455X (gfx1250) — matrix operand order repaired and checked by reading, not compiled, not run on hardware
//
#include <hip/hip_runtime.h>
#include <hip/hip_bf16.h>
#include <math.h>

typedef _Float16 f16;
typedef _Float16 v16h __attribute__((ext_vector_type(16)));
typedef _Float16 v8h  __attribute__((ext_vector_type(8)));
typedef float    v8f  __attribute__((ext_vector_type(8)));

#define C_V  50257
#define C_T  1024
#define C_D  768
#define C_H  12
#define C_L  6
#define C_B  2
#define C_FF 3072
#define C_HD 64
#define C_M  (C_B*C_T)   // 2048 rows

#define GEMM_GELU 1
#define GEMM_RES  2

__device__ __forceinline__ float gelu_ref(float x){
    // faithful to the reference's buggy NewGelu: tanh arg = c * 0.044715 * x^4
    const float c = 0.7978845608028654f;
    return 0.5f*x*(1.0f + tanhf(c*(x*0.044715f*x*x*x)));
}

// ------- weight convert + transpose: W[K][N] fp32 -> WT[N][K] f16 ----------
__global__ void k_cvtT(const float* __restrict__ src, f16* __restrict__ dst,
                       int K, int N){
    long long i = (long long)blockIdx.x*blockDim.x + threadIdx.x;
    long long tot = (long long)N*K;
    if (i >= tot) return;
    int c = (int)(i / K);               // output row  = n
    int r = (int)(i - (long long)c*K);  // output col  = k
    dst[i] = (f16)src[(long long)r*N + c];
}

// ------- embedding gather --------------------------------------------------
__global__ void k_embed(const int* __restrict__ ids, const float* __restrict__ wte,
                        const float* __restrict__ wpe, float* __restrict__ x){
    int i = blockIdx.x*blockDim.x + threadIdx.x;
    if (i >= C_M*C_D) return;
    int m = i / C_D, d = i - m*C_D;
    int t = m % C_T;
    int tok = ids[m];
    x[i] = wte[(long long)tok*C_D + d] + wpe[t*C_D + d];
}

// ------- split qkv -> q16 (x0.125), k16 [B,T,D] f16, vT [B,H,hd,T] f16 -----
__global__ void k_split(const float* __restrict__ qkv, f16* __restrict__ q16,
                        f16* __restrict__ k16, f16* __restrict__ vT){
    int i = blockIdx.x*blockDim.x + threadIdx.x;
    if (i >= C_M*C_D) return;
    int m = i / C_D, d = i - m*C_D;
    const float* row = qkv + (long long)m*3*C_D;
    q16[i] = (f16)(row[d] * 0.125f);          // 1/sqrt(64) folded into Q
    k16[i] = (f16)row[C_D + d];
    int b = m / C_T, t = m - b*C_T, h = d >> 6, hd = d & 63;
    vT[(((long long)(b*C_H + h))*C_HD + hd)*C_T + t] = (f16)row[2*C_D + d];
}

// ------- layernorm (+optional residual) ------------------------------------
__global__ __launch_bounds__(256) void k_ln(
    const float* __restrict__ in, const float* __restrict__ gg,
    const float* __restrict__ bb, const float* __restrict__ res,
    float* __restrict__ out, int hasres){
    __shared__ float s1[256], s2[256];
    int row = blockIdx.x, tid = threadIdx.x;
    const float* p = in + (long long)row*C_D;
    float a=0.f, b=0.f;
    for (int i=tid;i<C_D;i+=256){ float v=p[i]; a+=v; b+=v*v; }
    s1[tid]=a; s2[tid]=b; __syncthreads();
    for (int s=128;s>0;s>>=1){
        if (tid<s){ s1[tid]+=s1[tid+s]; s2[tid]+=s2[tid+s]; }
        __syncthreads();
    }
    float mu  = s1[0]*(1.f/C_D);
    float var = s2[0]*(1.f/C_D) - mu*mu;
    float rs  = rsqrtf(var + 1e-5f);
    for (int i=tid;i<C_D;i+=256){
        float v = (p[i]-mu)*rs*gg[i] + bb[i];
        if (hasres) v += res[(long long)row*C_D + i];
        out[(long long)row*C_D + i] = v;
    }
}

// ------- WMMA GEMM: C[M,N] = act(A[M,K] @ W + bias) (+Res), W given as WT[N][K]
// block = 256 threads = 8 waves, block tile 128x128, K step 32.
// LDS layouts chosen so EVERY fragment read is contiguous 16B -> ds_load_b128.
__global__ __launch_bounds__(256) void k_gemm(
    const float* __restrict__ A, const f16* __restrict__ WT,
    const float* __restrict__ bias, const float* __restrict__ Res,
    float* __restrict__ C, int K, int N, int flags)
{
    __shared__ __align__(16) f16 As[128*40];   // [m][k] rows of 32 k, stride 40
    __shared__ __align__(16) f16 Bs[128*40];   // [n][k] rows of 32 k, stride 40
    int tid  = threadIdx.x;
    int lane = tid & 31, w = tid >> 5;
    int wr = w >> 1, wc = w & 1;
    int bm = blockIdx.y * 128, bn = blockIdx.x * 128;
    int lhalf = lane >> 4, l15 = lane & 15;
    int koff  = lhalf * 8;

    v8f acc[2][4];
    for (int i=0;i<2;++i) for(int j=0;j<4;++j) for(int e=0;e<8;++e) acc[i][j][e]=0.f;

    int ar = tid >> 1, ac = (tid & 1) * 16;   // A staging: 2 thr/row, 16 floats
    int brow = tid >> 1, bkh = (tid & 1)*16;  // B staging: 2 thr/col, 16 halfs

    for (int k0 = 0; k0 < K; k0 += 32){
        { // stage A tile (fp32 -> f16), [m][k]
            const float4* src = (const float4*)(A + (long long)(bm+ar)*K + k0 + ac);
            f16* d = As + ar*40 + ac;
            #pragma unroll
            for (int v=0; v<4; ++v){
                float4 f = src[v];
                d[v*4+0]=(f16)f.x; d[v*4+1]=(f16)f.y; d[v*4+2]=(f16)f.z; d[v*4+3]=(f16)f.w;
            }
        }
        { // stage B tile from WT (f16), [n][k] -- contiguous both sides
            f16* d = Bs + brow*40 + bkh;
            if (bn + brow < N){
                const v8h* src = (const v8h*)(WT + (long long)(bn+brow)*K + k0 + bkh);
                v8h x0 = src[0], x1 = src[1];
                #pragma unroll
                for (int e=0;e<8;++e){ d[e]=x0[e]; d[8+e]=x1[e]; }
                if (k0 + 32 < K)   // hint next K tile -> global_prefetch_b8
                    __builtin_prefetch(WT + (long long)(bn+brow)*K + k0 + 32 + bkh, 0, 1);
            } else {
                #pragma unroll
                for (int e=0;e<16;++e) d[e]=(f16)0.f;
            }
        }
        __syncthreads();

        // A frag (ISA 16x32 f16): lane half 0 -> K 0-7 & 16-23; half 1 -> 8-15 & 24-31
        v16h a[2];
        #pragma unroll
        for (int mt=0; mt<2; ++mt){
            const f16* ap = As + (wr*32 + mt*16 + l15)*40;
            #pragma unroll
            for (int e=0;e<8;++e) a[mt][e]   = ap[koff + e];
            #pragma unroll
            for (int e=0;e<8;++e) a[mt][8+e] = ap[16 + koff + e];
        }
        // B frag: lane half 0 -> K 0-7 & 16-23; half 1 -> 8-15 & 24-31, N = lane&15
        #pragma unroll
        for (int nt=0; nt<4; ++nt){
            v16h b;
            const f16* bp = Bs + (wc*64 + nt*16 + l15)*40;
            #pragma unroll
            for (int e=0;e<8;++e){ b[e] = bp[koff + e]; b[8+e] = bp[16 + koff + e]; }
            #pragma unroll
            for (int mt=0; mt<2; ++mt)
                acc[mt][nt] = __builtin_amdgcn_wmma_f32_16x16x32_f16(
                    false, a[mt], false, b, (short)0, acc[mt][nt], false, false);
        }
        __syncthreads();
    }

    // epilogue: D layout -> row = g + 8*lhalf, col = lane&15
    #pragma unroll
    for (int mt=0; mt<2; ++mt)
      #pragma unroll
      for (int nt=0; nt<4; ++nt)
        #pragma unroll
        for (int g=0; g<8; ++g){
            int m = bm + wr*32 + mt*16 + g + 8*lhalf;
            int n = bn + wc*64 + nt*16 + l15;
            if (n >= N) continue;
            float v = acc[mt][nt][g] + bias[n];
            if (flags & GEMM_GELU) v = gelu_ref(v);
            if (flags & GEMM_RES)  v += Res[(long long)m*N + n];
            C[(long long)m*N + n] = v;
        }
}

// ------- attention: one wave per (b, h, 16-query tile); no causal mask -----
// P kept UNNORMALIZED in LDS (exp(s-max) <= 1); 1/sum folded into PV epilogue.
__global__ __launch_bounds__(32) void k_attn(
    const f16* __restrict__ q16, const f16* __restrict__ k16,
    const f16* __restrict__ vT, float* __restrict__ y){
    __shared__ __align__(16) f16 sc[16][1032];   // 16 x T probs, padded stride
    __shared__ __align__(16) float sinv[16];     // per-row 1/sum
    int lane = threadIdx.x;
    int q0 = blockIdx.x * 16, h = blockIdx.y, b = blockIdx.z;
    int lhalf = lane >> 4, l15 = lane & 15, koff = lhalf*8;

    // Q fragments (already scaled by 1/sqrt(64) in k_split)
    v16h aq[2];
    #pragma unroll
    for (int s=0; s<2; ++s){
        const f16* qp = q16 + (long long)(b*C_T + q0 + l15)*C_D + h*C_HD + s*32;
        #pragma unroll
        for (int e=0;e<8;++e) aq[s][e]   = qp[koff + e];
        #pragma unroll
        for (int e=0;e<8;++e) aq[s][8+e] = qp[16 + koff + e];
    }

    // scores S = Q @ K^T  (B frag: col = key token, K dim = head dim)
    for (int n0=0; n0<C_T; n0+=16){
        v8f acc; for (int e=0;e<8;++e) acc[e]=0.f;
        #pragma unroll
        for (int s=0; s<2; ++s){
            v16h bk;
            const f16* kp = k16 + (long long)(b*C_T + n0 + l15)*C_D
                          + h*C_HD + s*32;
            #pragma unroll
            for (int e=0;e<8;++e){ bk[e] = kp[koff + e]; bk[8+e] = kp[16 + koff + e]; }
            acc = __builtin_amdgcn_wmma_f32_16x16x32_f16(
                false, aq[s], false, bk, (short)0, acc, false, false);
        }
        #pragma unroll
        for (int g=0; g<8; ++g) sc[g + 8*lhalf][n0 + l15] = (f16)acc[g];
    }
    __syncthreads();

    // softmax (unnormalized): each lane owns half of row l15; vectorized v8h.
    {
        int r = l15;
        v8h* rowp = (v8h*)&sc[r][lhalf*512];   // 64 chunks of 8 halfs, 16B aligned
        float mx = -1e30f;
        for (int c=0;c<64;++c){
            v8h xv = rowp[c];
            #pragma unroll
            for (int e=0;e<8;++e) mx = fmaxf(mx, (float)xv[e]);
        }
        mx = fmaxf(mx, __shfl_xor(mx, 16, 32));
        float sum = 0.f;
        for (int c=0;c<64;++c){
            v8h xv = rowp[c], ov;
            #pragma unroll
            for (int e=0;e<8;++e){
                float ev = __expf((float)xv[e] - mx);
                sum += ev;
                ov[e] = (f16)ev;
            }
            rowp[c] = ov;
        }
        sum += __shfl_xor(sum, 16, 32);
        if (lhalf == 0) sinv[r] = 1.f/sum;
    }
    __syncthreads();

    // Y = (P_unnorm @ V) * (1/sum)  : all fragment reads contiguous 16B
    v8f acc[4];
    for (int nt=0;nt<4;++nt) for(int e=0;e<8;++e) acc[nt][e]=0.f;
    long long vbase = ((long long)(b*C_H + h))*C_HD*C_T;
    for (int k0=0; k0<C_T; k0+=32){
        v16h ap;
        #pragma unroll
        for (int e=0;e<8;++e) ap[e]   = sc[l15][k0 + koff + e];
        #pragma unroll
        for (int e=0;e<8;++e) ap[8+e] = sc[l15][k0 + 16 + koff + e];
        #pragma unroll
        for (int nt=0; nt<4; ++nt){
            v16h bv;
            const f16* vp = vT + vbase + (long long)(nt*16 + l15)*C_T
                          + k0;
            #pragma unroll
            for (int e=0;e<8;++e){ bv[e] = vp[koff + e]; bv[8+e] = vp[16 + koff + e]; }
            acc[nt] = __builtin_amdgcn_wmma_f32_16x16x32_f16(
                false, ap, false, bv, (short)0, acc[nt], false, false);
        }
    }
    // per-row scale: rows g + 8*lhalf -> two float4 LDS loads
    float invv[8];
    {
        const float4* ip = (const float4*)&sinv[8*lhalf];
        float4 i0 = ip[0], i1 = ip[1];
        invv[0]=i0.x; invv[1]=i0.y; invv[2]=i0.z; invv[3]=i0.w;
        invv[4]=i1.x; invv[5]=i1.y; invv[6]=i1.z; invv[7]=i1.w;
    }
    #pragma unroll
    for (int nt=0;nt<4;++nt)
        #pragma unroll
        for (int g=0; g<8; ++g)
            y[(long long)(b*C_T + q0 + g + 8*lhalf)*C_D + h*C_HD + nt*16 + l15]
                = acc[nt][g] * invv[g];
}

// ------- host orchestration ------------------------------------------------
extern "C" void kernel_launch(void* const* d_in, const int* in_sizes, int n_in,
                              void* d_out, int out_size, void* d_ws, size_t ws_size,
                              hipStream_t stream){
    (void)in_sizes; (void)n_in; (void)out_size; (void)ws_size;
    const int*   ids   = (const int*)d_in[0];
    const float* wte   = (const float*)d_in[1];
    const float* wpe   = (const float*)d_in[2];
    const float* Wqkv  = (const float*)d_in[3];
    const float* bqkv  = (const float*)d_in[4];
    const float* Wo    = (const float*)d_in[5];
    const float* bo    = (const float*)d_in[6];
    const float* ln1g  = (const float*)d_in[7];
    const float* ln1b  = (const float*)d_in[8];
    const float* W1    = (const float*)d_in[9];
    const float* b1    = (const float*)d_in[10];
    const float* W2    = (const float*)d_in[11];
    const float* b2    = (const float*)d_in[12];
    const float* lnfg  = (const float*)d_in[13];
    const float* lnfb  = (const float*)d_in[14];
    const float* Whead = (const float*)d_in[15];
    const float* bhead = (const float*)d_in[16];
    float* out = (float*)d_out;

    char* ws = (char*)d_ws;
    size_t off = 0;
    auto alloc = [&](size_t bytes)->char*{
        char* p = ws + off;
        off = (off + bytes + 255) & ~(size_t)255;
        return p;
    };
    float* x    = (float*)alloc((size_t)C_M*C_D*4);
    float* qkv  = (float*)alloc((size_t)C_M*3*C_D*4);
    float* yb   = (float*)alloc((size_t)C_M*C_D*4);
    float* pb   = (float*)alloc((size_t)C_M*C_D*4);
    float* hb   = (float*)alloc((size_t)C_M*C_FF*4);
    f16* q16    = (f16*)alloc((size_t)C_M*C_D*2);
    f16* kk16   = (f16*)alloc((size_t)C_M*C_D*2);
    f16* vT16   = (f16*)alloc((size_t)C_M*C_D*2);
    f16* w16qkv = (f16*)alloc((size_t)C_L*C_D*3*C_D*2);   // [l][n][k]
    f16* w16o   = (f16*)alloc((size_t)C_L*C_D*C_D*2);
    f16* w16f1  = (f16*)alloc((size_t)C_L*C_D*C_FF*2);
    f16* w16f2  = (f16*)alloc((size_t)C_L*C_FF*C_D*2);
    f16* w16h   = (f16*)alloc((size_t)C_D*(size_t)C_V*2);

    auto cvtT = [&](const float* s, f16* d, int K, int N){
        long long tot = (long long)N*K;
        k_cvtT<<<dim3((unsigned)((tot+255)/256)), 256, 0, stream>>>(s, d, K, N);
    };
    for (int l=0; l<C_L; ++l){
        cvtT(Wqkv + (size_t)l*C_D*3*C_D, w16qkv + (size_t)l*C_D*3*C_D, C_D, 3*C_D);
        cvtT(Wo   + (size_t)l*C_D*C_D,   w16o   + (size_t)l*C_D*C_D,   C_D, C_D);
        cvtT(W1   + (size_t)l*C_D*C_FF,  w16f1  + (size_t)l*C_D*C_FF,  C_D, C_FF);
        cvtT(W2   + (size_t)l*C_FF*C_D,  w16f2  + (size_t)l*C_FF*C_D,  C_FF, C_D);
    }
    cvtT(Whead, w16h, C_D, C_V);

    k_embed<<<(C_M*C_D+255)/256, 256, 0, stream>>>(ids, wte, wpe, x);

    auto gemm = [&](const float* A, const f16* WT, const float* bias, const float* Res,
                    float* C, int K, int N, int flags){
        dim3 grid((N + 127)/128, C_M/128);
        k_gemm<<<grid, 256, 0, stream>>>(A, WT, bias, Res, C, K, N, flags);
    };

    for (int l=0; l<C_L; ++l){
        gemm(x,  w16qkv + (size_t)l*C_D*3*C_D, bqkv + l*3*C_D, nullptr, qkv,
             C_D, 3*C_D, 0);
        k_split<<<(C_M*C_D+255)/256, 256, 0, stream>>>(qkv, q16, kk16, vT16);
        k_attn<<<dim3(C_T/16, C_H, C_B), 32, 0, stream>>>(q16, kk16, vT16, yb);
        gemm(yb, w16o + (size_t)l*C_D*C_D,     bo + l*C_D,     nullptr, pb,
             C_D, C_D, 0);
        // x = LN(attn_proj) + x   (reference applies ln1 AFTER attention)
        k_ln<<<C_M, 256, 0, stream>>>(pb, ln1g + l*C_D, ln1b + l*C_D, x, x, 1);
        gemm(x,  w16f1 + (size_t)l*C_D*C_FF,   b1 + l*C_FF,    nullptr, hb,
             C_D, C_FF, GEMM_GELU);
        gemm(hb, w16f2 + (size_t)l*C_FF*C_D,   b2 + l*C_D,     x,       x,
             C_FF, C_D, GEMM_RES);
    }
    k_ln<<<C_M, 256, 0, stream>>>(x, lnfg, lnfb, nullptr, yb, 0);
    gemm(yb, w16h, bhead, nullptr, out, C_D, C_V, 0);
}
